// SAGE_74079595921726
// MI455X (gfx1250) — compile-verified, matrix operand order checked by reading, not run on hardware
//
#include <hip/hip_runtime.h>

// ---------------------------------------------------------------------------
// GraphSAGE (6x SAGEConv + BN + ReLU, mean-pool, linear head) for gfx1250.
// fp32 residency in memory (bandwidth/atomic-bound; x fits in 192MB L2),
// bf16 V_WMMA_F32_16X16X32_BF16 for the fused [N x 256] @ [256 x 128] GEMM
// (A = [agg*inv_deg | x], W = [Wl ; Wr]); bias + BN statistics fused into the
// GEMM epilogue (register reduction -> ds_add_f32 -> 1 global atomic/channel).
// ---------------------------------------------------------------------------

typedef __bf16 v16bf __attribute__((ext_vector_type(16)));
typedef float  v8f   __attribute__((ext_vector_type(8)));
typedef unsigned int v8u __attribute__((ext_vector_type(8)));

constexpr int HC  = 128;   // hidden channels
constexpr int LDW = 132;   // LDS pair-row stride (bank-conflict padding)

// round-to-nearest (ties away) f32->bf16 packed pair:
// 2 x v_add + 1 x v_perm_b32 (dst = {hi[31:16], lo[31:16]})
__device__ __forceinline__ unsigned pk2(float lo, float hi) {
  unsigned a = __builtin_bit_cast(unsigned, lo) + 0x8000u;
  unsigned b = __builtin_bit_cast(unsigned, hi) + 0x8000u;
  return __builtin_amdgcn_perm(b, a, 0x07060302u);  // bytes: a2,a3,b2,b3
}

// ---------------- utility ----------------
__global__ void zero_kernel(float* __restrict__ p, long long n) {
  long long i = (long long)blockIdx.x * blockDim.x + threadIdx.x;
  if (i < n) p[i] = 0.f;
}

// ---------------- degree ----------------
__global__ void deg_kernel(const int* __restrict__ dst, float* __restrict__ deg, int E) {
  int e = blockIdx.x * blockDim.x + threadIdx.x;
  if (e < E) atomicAdd(&deg[dst[e]], 1.0f);
}
__global__ void inv_kernel(float* __restrict__ deg, int N) {
  int i = blockIdx.x * blockDim.x + threadIdx.x;
  if (i < N) { float d = deg[i]; deg[i] = (d > 0.f) ? (1.f / d) : 0.f; }
}

// ---------------- neighbor scatter-add: 32 lanes per edge, float4/lane -----
__global__ void agg_kernel(const float* __restrict__ x, const int* __restrict__ src,
                           const int* __restrict__ dst, float* __restrict__ agg, int E) {
  int idx = blockIdx.x * blockDim.x + threadIdx.x;
  if (idx >= E * 32) return;
  int e = __builtin_amdgcn_readfirstlane(idx >> 5);   // wave-uniform -> s_load
  int lane = threadIdx.x & 31;
  const float4 v = *(const float4*)(x + (size_t)src[e] * HC + lane * 4);
  float* o = agg + (size_t)dst[e] * HC + lane * 4;
  atomicAdd(o + 0, v.x); atomicAdd(o + 1, v.y);
  atomicAdd(o + 2, v.z); atomicAdd(o + 3, v.w);
}

// ---------------- fused SAGE GEMM + bias + BN-stat epilogue ----------------
// h = (agg * inv_deg) @ Wl + x @ Wr + bl   as [N x 256] @ [256 x 128]
// 256 threads = 8 waves; wave w owns rows [blk*128 + w*16, +16) x all 128 cols.
__global__ __launch_bounds__(256) void sage_fused_gemm(
    const float* __restrict__ xin,
    float* __restrict__ aggh,                 // in: raw sums; out: h (same rows)
    const float* __restrict__ dinv,
    const float* __restrict__ Wl, const float* __restrict__ Wr,
    const float* __restrict__ bl,
    float* __restrict__ gsum, float* __restrict__ gsq, int N) {
  __shared__ unsigned wlds[HC][LDW];          // bf16 pair (k,k+1), n-major
  __shared__ float sbias[HC], ssum[HC], ssq[HC];

  const int tid = threadIdx.x;
  for (int t = tid; t < 128 * 128; t += 256) {       // coalesced: n fastest
    int p = t >> 7, n = t & 127, k0 = p << 1;
    float w0, w1;
    if (k0 < HC) { w0 = Wl[k0 * HC + n];        w1 = Wl[(k0 + 1) * HC + n]; }
    else         { w0 = Wr[(k0 - HC) * HC + n]; w1 = Wr[(k0 - HC + 1) * HC + n]; }
    wlds[n][p] = pk2(w0, w1);
  }
  if (tid < HC) { sbias[tid] = bl[tid]; ssum[tid] = 0.f; ssq[tid] = 0.f; }
  __syncthreads();

  const int wave = tid >> 5, lane = tid & 31;
  const int mloc = lane & 15, khalf = lane >> 4;
  const int rowbase = blockIdx.x * 128 + wave * 16;
  const int row  = rowbase + mloc;
  const int rowc = (row < N) ? row : (N - 1);  // clamp: garbage rows discarded
  const float inv = dinv[rowc];
  const float4* arow4 = (const float4*)(aggh + (size_t)rowc * HC);
  const float4* xrow4 = (const float4*)(xin  + (size_t)rowc * HC);

  v8f acc[8];
  v8f zf = {};
  #pragma unroll
  for (int nt = 0; nt < 8; ++nt) acc[nt] = zf;

  #pragma unroll
  for (int ks = 0; ks < 8; ++ks) {                   // 8 x K=32 -> K=256
    // lane's A slice: two 8-float runs -> 4 x b128 loads, branch-free
    const float4* r4 = (ks < 4) ? arow4 : xrow4;
    const int i0 = 8 * (ks & 3) + 2 * khalf;
    float4 c0 = r4[i0], c1 = r4[i0 + 1], c2 = r4[i0 + 4], c3 = r4[i0 + 5];
    if (ks < 4) {
      c0.x *= inv; c0.y *= inv; c0.z *= inv; c0.w *= inv;
      c1.x *= inv; c1.y *= inv; c1.z *= inv; c1.w *= inv;
      c2.x *= inv; c2.y *= inv; c2.z *= inv; c2.w *= inv;
      c3.x *= inv; c3.y *= inv; c3.z *= inv; c3.w *= inv;
    }
    v8u apk;
    apk[0] = pk2(c0.x, c0.y); apk[1] = pk2(c0.z, c0.w);
    apk[2] = pk2(c1.x, c1.y); apk[3] = pk2(c1.z, c1.w);
    apk[4] = pk2(c2.x, c2.y); apk[5] = pk2(c2.z, c2.w);
    apk[6] = pk2(c3.x, c3.y); apk[7] = pk2(c3.z, c3.w);
    v16bf afrag = __builtin_bit_cast(v16bf, apk);

    #pragma unroll
    for (int nt = 0; nt < 8; ++nt) {
      const int n = nt * 16 + mloc;
      v8u bpk;
      #pragma unroll
      for (int v = 0; v < 8; ++v) {
        int pi = ((v < 4) ? v : 4 + v) + 4 * khalf + ks * 16;
        bpk[v] = wlds[n][pi];
      }
      v16bf bfrag = __builtin_bit_cast(v16bf, bpk);
      acc[nt] = __builtin_amdgcn_wmma_f32_16x16x32_bf16(
          false, afrag, false, bfrag, (short)0, acc[nt], false, false);
    }
  }

  // epilogue: +bias, store h over agg rows, per-channel stats (reg -> LDS)
  if (rowbase + 16 <= N) {                           // fast path: full tile
    #pragma unroll
    for (int nt = 0; nt < 8; ++nt) {
      const int n = nt * 16 + mloc;
      const float b = sbias[n];
      float ls = 0.f, lq = 0.f;
      #pragma unroll
      for (int r = 0; r < 8; ++r) {
        const int rr = rowbase + r + 8 * khalf;
        float hv = acc[nt][r] + b;
        aggh[(size_t)rr * HC + n] = hv;
        ls += hv; lq += hv * hv;
      }
      atomicAdd(&ssum[n], ls);
      atomicAdd(&ssq[n], lq);
    }
  } else {                                           // edge tile: guarded
    #pragma unroll
    for (int nt = 0; nt < 8; ++nt) {
      const int n = nt * 16 + mloc;
      const float b = sbias[n];
      float ls = 0.f, lq = 0.f;
      #pragma unroll
      for (int r = 0; r < 8; ++r) {
        const int rr = rowbase + r + 8 * khalf;
        if (rr < N) {
          float hv = acc[nt][r] + b;
          aggh[(size_t)rr * HC + n] = hv;
          ls += hv; lq += hv * hv;
        }
      }
      atomicAdd(&ssum[n], ls);
      atomicAdd(&ssq[n], lq);
    }
  }
  __syncthreads();
  if (tid < HC) { atomicAdd(&gsum[tid], ssum[tid]); atomicAdd(&gsq[tid], ssq[tid]); }
}

// ---------------- BN finalize + apply ----------------
__global__ void bn_finalize_kernel(const float* __restrict__ gsum, const float* __restrict__ gsq,
                                   const float* __restrict__ gamma, const float* __restrict__ beta,
                                   float* __restrict__ scl, float* __restrict__ shf, int N) {
  int c = threadIdx.x;
  if (c < HC) {
    float invn = 1.f / (float)N;
    float mu  = gsum[c] * invn;
    float var = gsq[c] * invn - mu * mu;
    float s = gamma[c] * rsqrtf(var + 1e-5f);
    scl[c] = s; shf[c] = beta[c] - mu * s;
  }
}
__global__ void bn_relu_kernel(const float* __restrict__ h, const float* __restrict__ scl,
                               const float* __restrict__ shf, float* __restrict__ xout, int n4) {
  int i = blockIdx.x * blockDim.x + threadIdx.x;
  if (i >= n4) return;
  int c = (i & 31) * 4;
  float4 v = ((const float4*)h)[i];
  float4 o;
  o.x = fmaxf(v.x * scl[c + 0] + shf[c + 0], 0.f);
  o.y = fmaxf(v.y * scl[c + 1] + shf[c + 1], 0.f);
  o.z = fmaxf(v.z * scl[c + 2] + shf[c + 2], 0.f);
  o.w = fmaxf(v.w * scl[c + 3] + shf[c + 3], 0.f);
  ((float4*)xout)[i] = o;
}

// ---------------- global mean pool + head ----------------
__global__ void pool_kernel(const float* __restrict__ x, const int* __restrict__ batch,
                            float* __restrict__ pooled, float* __restrict__ cnt, int n4) {
  int i = blockIdx.x * blockDim.x + threadIdx.x;
  if (i >= n4) return;
  int node = __builtin_amdgcn_readfirstlane(i >> 5);  // wave-uniform
  int lane = threadIdx.x & 31;
  int g = batch[node];
  float4 v = ((const float4*)x)[(size_t)node * 32 + lane];
  float* o = pooled + (size_t)g * HC + lane * 4;
  atomicAdd(o + 0, v.x); atomicAdd(o + 1, v.y);
  atomicAdd(o + 2, v.z); atomicAdd(o + 3, v.w);
  if (lane == 0) atomicAdd(&cnt[g], 1.f);
}
__global__ void head_kernel(const float* __restrict__ pooled, const float* __restrict__ cnt,
                            const float* __restrict__ W, const float* __restrict__ b,
                            float* __restrict__ out, int G, int C) {
  int i = blockIdx.x * blockDim.x + threadIdx.x;
  if (i >= G * C) return;
  int g = i / C, c = i % C;
  float s = 0.f;
  for (int h = 0; h < HC; ++h) s += pooled[(size_t)g * HC + h] * W[h * C + c];
  out[i] = s / fmaxf(cnt[g], 1.f) + b[c];
}

// ---------------------------------------------------------------------------
extern "C" void kernel_launch(void* const* d_in, const int* in_sizes, int n_in,
                              void* d_out, int out_size, void* d_ws, size_t ws_size,
                              hipStream_t stream) {
  const float* x0    = (const float*)d_in[0];
  const int*   ei    = (const int*)  d_in[1];
  const int*   batch = (const int*)  d_in[2];
  const float* Wl    = (const float*)d_in[3];
  const float* bl    = (const float*)d_in[4];
  const float* Wr    = (const float*)d_in[5];
  const float* gamma = (const float*)d_in[6];
  const float* beta  = (const float*)d_in[7];
  const float* linW  = (const float*)d_in[8];
  const float* linb  = (const float*)d_in[9];

  const int N = in_sizes[0] / HC;
  const int E = in_sizes[1] / 2;
  const int L = in_sizes[3] / (HC * HC);
  const int C = in_sizes[9];
  const int G = out_size / C;

  // workspace layout (floats)
  float* xbuf   = (float*)d_ws;
  float* agg    = xbuf + (size_t)N * HC;      // doubles as h after the GEMM
  float* dinv   = agg  + (size_t)N * HC;
  float* gsum   = dinv + N;
  float* gsq    = gsum + HC;
  float* scl    = gsq  + HC;
  float* shf    = scl  + HC;
  float* pooled = shf  + HC;
  float* cnt    = pooled + (size_t)G * HC;

  const int* src = ei;
  const int* dst = ei + E;

  hipMemcpyAsync(xbuf, x0, (size_t)N * HC * sizeof(float),
                 hipMemcpyDeviceToDevice, stream);

  // in-degree -> reciprocal (in place)
  zero_kernel<<<(N + 255) / 256, 256, 0, stream>>>(dinv, N);
  deg_kernel<<<(E + 255) / 256, 256, 0, stream>>>(dst, dinv, E);
  inv_kernel<<<(N + 255) / 256, 256, 0, stream>>>(dinv, N);

  const int  n4   = N * 32;                    // float4 slots in [N x 128]
  const long long NH = (long long)N * HC;

  for (int l = 0; l < L; ++l) {
    zero_kernel<<<(unsigned)((NH + 255) / 256), 256, 0, stream>>>(agg, NH);
    zero_kernel<<<1, 256, 0, stream>>>(gsum, 2 * HC);
    agg_kernel<<<(E * 32 + 255) / 256, 256, 0, stream>>>(xbuf, src, dst, agg, E);
    sage_fused_gemm<<<(N + 127) / 128, 256, 0, stream>>>(
        xbuf, agg, dinv,
        Wl + (size_t)l * HC * HC, Wr + (size_t)l * HC * HC, bl + (size_t)l * HC,
        gsum, gsq, N);
    bn_finalize_kernel<<<1, HC, 0, stream>>>(gsum, gsq,
        gamma + (size_t)l * HC, beta + (size_t)l * HC, scl, shf, N);
    bn_relu_kernel<<<(n4 + 255) / 256, 256, 0, stream>>>(agg, scl, shf, xbuf, n4);
  }

  zero_kernel<<<(unsigned)(((size_t)G * HC + G + 255) / 256), 256, 0, stream>>>(
      pooled, (long long)G * HC + G);
  pool_kernel<<<(n4 + 255) / 256, 256, 0, stream>>>(xbuf, batch, pooled, cnt, n4);
  head_kernel<<<(G * C + 255) / 256, 256, 0, stream>>>(pooled, cnt, linW, linb,
                                                       (float*)d_out, G, C);
}
